// additiveAttention_71975061947015
// MI455X (gfx1250) — compile-verified, matrix operand order checked by reading, not run on hardware
//
#include <hip/hip_runtime.h>
#include <hip/hip_bf16.h>

// ---------------------------------------------------------------------------
// Additive (Bahdanau) attention for MI455X / gfx1250, wave32 + WMMA bf16.
// B=4, L=256, HID=512, HEADS=8, DH=64.
// ---------------------------------------------------------------------------

#define B_    4
#define L_    256
#define HID_  512
#define H_    8
#define DH_   64
#define M_    (B_ * L_)          // 1024 rows of activations
#define OUT_X (B_ * L_ * HID_)   // 524288 floats: first output (out)
// attention output follows at d_out + OUT_X : [B,H,Lq,Lk] f32

typedef __attribute__((ext_vector_type(16))) __bf16 bf16x16;
typedef __attribute__((ext_vector_type(8)))  float  f32x8;

// Load one lane's A/B^T operand slice for v_wmma_f32_16x16x32_bf16 from a
// row-major matrix (row = M or N index, col = K index).
// Lane r (=lane&15) holds row r; half (=lane>>4) selects K {0..7,16..23} vs
// {8..15,24..31}. Two 16-byte contiguous loads per lane.
__device__ inline bf16x16 ld_op16(const __bf16* p) {
  union { bf16x16 v; uint4 q[2]; } u;
  u.q[0] = *(const uint4*)(p);
  u.q[1] = *(const uint4*)(p + 16);
  return u.v;
}

__device__ inline float fast_tanh(float x) {
  // tanh(x) = 1 - 2/(exp(2x)+1), exp via v_exp_f32 (exp2)
  float t = __builtin_amdgcn_exp2f(x * 2.885390081777927f); // 2*log2(e)
  return 1.0f - 2.0f * __builtin_amdgcn_rcpf(t + 1.0f);
}

__device__ inline float fast_exp(float x) {
  return __builtin_amdgcn_exp2f(x * 1.4426950408889634f);
}

// ---------------------------------------------------------------------------
// Fold the per-head DH x DH transform Wh into the big projection:
//   Weff[h*64+o, i] = sum_m Wh[o,m] * Wbig[h*64+m, i]   (bf16 out)
//   beff[h*64+o]    = sum_m Wh[o,m] * bbig[h*64+m] + bh[o]
// ---------------------------------------------------------------------------
__global__ __launch_bounds__(256)
void fuse_weight_kernel(const float* __restrict__ Wbig, const float* __restrict__ bbig,
                        const float* __restrict__ Wh,   const float* __restrict__ bh,
                        __bf16* __restrict__ Wout, float* __restrict__ bout) {
  int idx = blockIdx.x * 256 + threadIdx.x;
  if (idx < HID_ * HID_) {
    int o = idx >> 9;          // output channel 0..511
    int i = idx & 511;         // input channel
    int hh = o >> 6, oo = o & 63;
    float acc = 0.f;
    #pragma unroll 8
    for (int m = 0; m < DH_; ++m)
      acc += Wh[oo * DH_ + m] * Wbig[((size_t)hh * DH_ + m) * HID_ + i];
    Wout[(size_t)o * HID_ + i] = (__bf16)acc;
  } else if (idx < HID_ * HID_ + HID_) {
    int o = idx - HID_ * HID_;
    int hh = o >> 6, oo = o & 63;
    float acc = bh[oo];
    #pragma unroll 8
    for (int m = 0; m < DH_; ++m)
      acc += Wh[oo * DH_ + m] * bbig[hh * DH_ + m];
    bout[o] = acc;
  }
}

__global__ __launch_bounds__(256)
void cvt_bf16_kernel(const float* __restrict__ src, __bf16* __restrict__ dst, int n) {
  int i = blockIdx.x * 256 + threadIdx.x;
  if (i < n) dst[i] = (__bf16)src[i];
}

// ---------------------------------------------------------------------------
// Projection GEMM: Y[m,n] = sum_k A[m,k] * W[n,k] + bias[n]
//   A: [1024 x 512] bf16 row-major, W: [512 x 512] bf16 row-major (acts as B^T)
// Each wave owns a 16x32 output slab (one M-tile x two N-tiles): the A
// fragment is loaded once per K-step and reused by two WMMAs, halving A-side
// VMEM traffic. 8 waves/block, 128 blocks -> 1024 wave-tasks.
// mode 0: f32 out, per-head layout [b][h][l][d]        (Qt / Kt)
// mode 1: bf16 out, transposed per-head [b][h][d][l]   (V for the P@V GEMM)
// mode 2: f32 out, plain [m][n] row-major              (final projection)
// ---------------------------------------------------------------------------
__global__ __launch_bounds__(256)
void gemm_proj_kernel(const __bf16* __restrict__ A, const __bf16* __restrict__ W,
                      const float* __restrict__ bias,
                      float* __restrict__ outF, __bf16* __restrict__ outB, int mode) {
  int tid  = threadIdx.x;
  int lane = tid & 31, wid = tid >> 5;
  int t   = blockIdx.x * 8 + wid;   // 0..1023 (64 M-tiles x 16 N-pairs)
  int tm  = t >> 4;                 // M-tile 0..63
  int tnp = t & 15;                 // N-pair 0..15 -> columns tnp*32 ..
  int r = lane & 15, hf = lane >> 4;

  const __bf16* Ap  = A + ((size_t)tm * 16 + r) * HID_ + hf * 8;
  const __bf16* Bp0 = W + ((size_t)tnp * 32 + r) * HID_ + hf * 8;
  const __bf16* Bp1 = Bp0 + (size_t)16 * HID_;

  f32x8 acc0 = {}, acc1 = {};
  #pragma unroll 4
  for (int k = 0; k < HID_; k += 32) {
    // speculative L2 prefetch a few steps ahead (silently dropped if OOB)
    __builtin_prefetch(Ap  + k + 128, 0, 1);
    __builtin_prefetch(Bp0 + k + 128, 0, 1);
    __builtin_prefetch(Bp1 + k + 128, 0, 1);
    bf16x16 a  = ld_op16(Ap  + k);
    bf16x16 b0 = ld_op16(Bp0 + k);
    bf16x16 b1 = ld_op16(Bp1 + k);
    acc0 = __builtin_amdgcn_wmma_f32_16x16x32_bf16(false, a, false, b0,
                                                   (short)0, acc0, false, false);
    acc1 = __builtin_amdgcn_wmma_f32_16x16x32_bf16(false, a, false, b1,
                                                   (short)0, acc1, false, false);
  }

  int mBase = tm * 16 + hf * 8;     // lanes 0-15: M=v ; lanes 16-31: M=v+8
  #pragma unroll
  for (int half = 0; half < 2; ++half) {
    int n  = tnp * 32 + half * 16 + (lane & 15);
    float bv = bias[n];
    const f32x8& acc = half ? acc1 : acc0;
    #pragma unroll
    for (int v = 0; v < 8; ++v) {
      int m = mBase + v;
      float val = acc[v] + bv;
      if (mode == 0) {
        int bb = m >> 8, l = m & 255, hh = n >> 6, d = n & 63;
        outF[(((size_t)bb * H_ + hh) * L_ + l) * DH_ + d] = val;
      } else if (mode == 1) {
        int bb = m >> 8, l = m & 255, hh = n >> 6, d = n & 63;
        outB[(((size_t)bb * H_ + hh) * DH_ + d) * L_ + l] = (__bf16)val;
      } else {
        outF[(size_t)m * HID_ + n] = val;
      }
    }
  }
}

// ---------------------------------------------------------------------------
// Energy + softmax + P@V. One block = one (b,h) pair and 32 query rows.
// grid = 32 bh * 8 qtiles = 256 blocks, 256 threads (8 wave32).
//   e[q,k] = sum_d va[d]*tanh(Qt[q,d]+Kt[k,d]) + va_b   (VALU/TRANS work)
//   P = softmax_k(e)  -> fp32 to attention output, bf16 to LDS
//   X[:, h*64:...] = P @ V                              (WMMA from LDS)
// ---------------------------------------------------------------------------
#define LQT 32

__global__ __launch_bounds__(256)
void energy_softmax_pv_kernel(const float* __restrict__ Qt, const float* __restrict__ Kt,
                              const __bf16* __restrict__ Vt,
                              const float* __restrict__ va_w, const float* __restrict__ va_b,
                              float* __restrict__ attn, __bf16* __restrict__ Xc) {
  __shared__ float  sK[L_ * DH_];                                   // 64 KB
  __shared__ float  sQt[DH_ * LQT];                                 // 8 KB  (transposed [d][q])
  __shared__ float  sE[LQT * L_];                                   // 32 KB
  __shared__ __attribute__((aligned(16))) __bf16 sPb[LQT * L_];     // 16 KB
  __shared__ float  sVa[DH_];
  __shared__ float  sRed[LQT * 8];
  __shared__ float  sRowM[LQT], sRowS[LQT];

  int tid = threadIdx.x;
  int bh  = blockIdx.x >> 3;           // 0..31
  int qt  = blockIdx.x & 7;
  int q0  = qt * LQT;
  int b   = bh >> 3, h = bh & 7;

  const float* KtB = Kt + (size_t)bh * L_ * DH_;
  const float* QtB = Qt + (size_t)bh * L_ * DH_ + (size_t)q0 * DH_;

  // Stage Kt tile (16384 floats) as float4
  {
    const float4* src = (const float4*)KtB;
    float4* dst = (float4*)sK;
    for (int i = tid; i < (L_ * DH_) / 4; i += 256) dst[i] = src[i];
  }
  // Stage Qt transposed: sQt[d][q] so lane-per-q reads are conflict-free
  for (int i = tid; i < LQT * DH_; i += 256) {
    int q = i >> 6, d = i & 63;
    sQt[d * LQT + q] = QtB[i];
  }
  if (tid < DH_) sVa[tid] = va_w[tid];
  __syncthreads();

  float vab = va_b[0];
  int lane = tid & 31;     // lane == query row within tile
  int wid  = tid >> 5;

  // energy: each wave sweeps k = wid, wid+8, ... ; sK[k][*] is a broadcast
  for (int kk = wid; kk < L_; kk += 8) {
    const float* kp = sK + kk * DH_;
    float e = vab;
    #pragma unroll 8
    for (int d = 0; d < DH_; ++d) {
      float x = sQt[d * LQT + lane] + kp[d];
      e += sVa[d] * fast_tanh(x);
    }
    sE[lane * L_ + kk] = e;
  }
  __syncthreads();

  // softmax over k: 8 threads per row, 32 columns each
  int row = tid >> 3, c = tid & 7;
  {
    float m = -1e30f;
    for (int j = 0; j < 32; ++j) m = fmaxf(m, sE[row * L_ + c * 32 + j]);
    sRed[row * 8 + c] = m;
  }
  __syncthreads();
  if (c == 0) {
    float mm = sRed[row * 8];
    for (int j = 1; j < 8; ++j) mm = fmaxf(mm, sRed[row * 8 + j]);
    sRowM[row] = mm;
  }
  __syncthreads();
  {
    float rm = sRowM[row];
    float s = 0.f;
    for (int j = 0; j < 32; ++j) {
      float p = fast_exp(sE[row * L_ + c * 32 + j] - rm);
      sE[row * L_ + c * 32 + j] = p;
      s += p;
    }
    sRed[row * 8 + c] = s;
  }
  __syncthreads();
  if (c == 0) {
    float ss = sRed[row * 8];
    for (int j = 1; j < 8; ++j) ss += sRed[row * 8 + j];
    sRowS[row] = __builtin_amdgcn_rcpf(ss);
  }
  __syncthreads();
  {
    float rs = sRowS[row];
    float* attnRow = attn + ((size_t)bh * L_ + q0 + row) * L_;
    for (int j = 0; j < 32; ++j) {
      int col = c * 32 + j;
      float p = sE[row * L_ + col] * rs;
      attnRow[col] = p;                    // attention output (fp32)
      sPb[row * L_ + col] = (__bf16)p;     // WMMA A operand (bf16)
    }
  }
  __syncthreads();

  // P @ V : M=32 (2 tiles) x N=64 (4 tiles), K=256. One tile per wave.
  {
    int tmm = wid >> 2, tnn = wid & 3;
    int r = lane & 15, hf = lane >> 4;
    f32x8 acc = {};
    const __bf16* Ap = sPb + (tmm * 16 + r) * L_ + hf * 8;
    const __bf16* Bp = Vt + ((size_t)bh * DH_ + tnn * 16 + r) * L_ + hf * 8;
    #pragma unroll 2
    for (int k = 0; k < L_; k += 32) {
      bf16x16 a = ld_op16(Ap + k);
      bf16x16 bmat = ld_op16(Bp + k);
      acc = __builtin_amdgcn_wmma_f32_16x16x32_bf16(false, a, false, bmat,
                                                    (short)0, acc, false, false);
    }
    int nCol  = h * DH_ + tnn * 16 + (lane & 15);
    int mBase = q0 + tmm * 16 + hf * 8;
    #pragma unroll
    for (int v = 0; v < 8; ++v) {
      int l = mBase + v;
      Xc[((size_t)b * L_ + l) * HID_ + nCol] = (__bf16)acc[v];
    }
  }
}

// ---------------------------------------------------------------------------
extern "C" void kernel_launch(void* const* d_in, const int* in_sizes, int n_in,
                              void* d_out, int out_size, void* d_ws, size_t ws_size,
                              hipStream_t stream) {
  const float* query = (const float*)d_in[0];
  const float* key_  = (const float*)d_in[1];
  const float* value = (const float*)d_in[2];
  const float* Wq = (const float*)d_in[3];  const float* bq = (const float*)d_in[4];
  const float* Wk = (const float*)d_in[5];  const float* bk = (const float*)d_in[6];
  const float* Wv = (const float*)d_in[7];  const float* bv = (const float*)d_in[8];
  const float* W1 = (const float*)d_in[9];  const float* b1 = (const float*)d_in[10];
  const float* W2 = (const float*)d_in[11]; const float* b2 = (const float*)d_in[12];
  const float* va_w = (const float*)d_in[13];
  const float* va_b = (const float*)d_in[14];
  const float* Wo = (const float*)d_in[15]; const float* bo = (const float*)d_in[16];

  float* out  = (float*)d_out;
  float* attn = out + OUT_X;

  // workspace carve-out (~11.1 MB)
  char* p = (char*)d_ws;
  auto carve = [&](size_t bytes) { void* r = (void*)p; p += (bytes + 255) & ~(size_t)255; return r; };
  __bf16* WqE = (__bf16*)carve(HID_ * HID_ * 2);
  __bf16* WkE = (__bf16*)carve(HID_ * HID_ * 2);
  __bf16* WvB = (__bf16*)carve(HID_ * HID_ * 2);
  __bf16* WoB = (__bf16*)carve(HID_ * HID_ * 2);
  float*  bqE = (float*)carve(HID_ * 4);
  float*  bkE = (float*)carve(HID_ * 4);
  __bf16* QB  = (__bf16*)carve((size_t)M_ * HID_ * 2);
  __bf16* KB  = (__bf16*)carve((size_t)M_ * HID_ * 2);
  __bf16* VB  = (__bf16*)carve((size_t)M_ * HID_ * 2);
  float*  Qt  = (float*)carve((size_t)M_ * HID_ * 4);
  float*  Kt  = (float*)carve((size_t)M_ * HID_ * 4);
  __bf16* Vt  = (__bf16*)carve((size_t)M_ * HID_ * 2);
  __bf16* Xc  = (__bf16*)carve((size_t)M_ * HID_ * 2);

  // 1) fold W1/W2 into Wq/Wk (bf16) + fused biases
  int fuseBlocks = (HID_ * HID_ + HID_ + 255) / 256;
  fuse_weight_kernel<<<fuseBlocks, 256, 0, stream>>>(Wq, bq, W1, b1, WqE, bqE);
  fuse_weight_kernel<<<fuseBlocks, 256, 0, stream>>>(Wk, bk, W2, b2, WkE, bkE);

  // 2) bf16 conversions of remaining weights + activations
  cvt_bf16_kernel<<<(HID_ * HID_ + 255) / 256, 256, 0, stream>>>(Wv, WvB, HID_ * HID_);
  cvt_bf16_kernel<<<(HID_ * HID_ + 255) / 256, 256, 0, stream>>>(Wo, WoB, HID_ * HID_);
  cvt_bf16_kernel<<<(M_ * HID_ + 255) / 256, 256, 0, stream>>>(query, QB, M_ * HID_);
  cvt_bf16_kernel<<<(M_ * HID_ + 255) / 256, 256, 0, stream>>>(key_,  KB, M_ * HID_);
  cvt_bf16_kernel<<<(M_ * HID_ + 255) / 256, 256, 0, stream>>>(value, VB, M_ * HID_);

  // 3) fused projections (WMMA bf16): Qt, Kt per-head f32; V transposed bf16
  gemm_proj_kernel<<<128, 256, 0, stream>>>(QB, WqE, bqE, Qt, nullptr, 0);
  gemm_proj_kernel<<<128, 256, 0, stream>>>(KB, WkE, bkE, Kt, nullptr, 0);
  gemm_proj_kernel<<<128, 256, 0, stream>>>(VB, WvB, bv, nullptr, Vt, 1);

  // 4) additive energy + softmax + P@V (writes attention output and Xc)
  energy_softmax_pv_kernel<<<256, 256, 0, stream>>>(Qt, Kt, Vt, va_w, va_b, attn, Xc);

  // 5) output projection (WMMA bf16) -> out
  gemm_proj_kernel<<<128, 256, 0, stream>>>(Xc, WoB, bo, out, nullptr, 2);
}
